// TransformerNet_16381005267485
// MI455X (gfx1250) — matrix operand order repaired and checked by reading, not compiled, not run on hardware
//
#include <hip/hip_runtime.h>

// ---------- types ----------
typedef __attribute__((ext_vector_type(16))) __bf16 v16bf;
typedef __attribute__((ext_vector_type(8)))  float  v8f;
typedef unsigned short u16;
typedef unsigned int   u32;
typedef unsigned long long u64;
typedef __attribute__((ext_vector_type(4))) unsigned int tdm_vu4;
typedef __attribute__((ext_vector_type(8))) int          tdm_vi8;
typedef __attribute__((ext_vector_type(4))) int          tdm_vi4;

// ---------- model constants ----------
constexpr int Bn = 32, CIN = 16, L0 = 3750;
constexpr int H = 256, NH = 8, DK = 32, DV = 32, INNER = 128, OUTD = 64;
constexpr int L1 = 1874;           // conv1: 3749 valid -> pool 1874
constexpr int L2 = 936;            // conv2: 1873 valid -> pool 936
constexpr int LSEQ = 467;          // conv3: 935 valid  -> pool 467
constexpr int BL = Bn * LSEQ;      // 14944 tokens (= 32 * 467, 32-divisible)
constexpr int SPAD = 480;          // padded key length (mult of 32)

// ---------- bf16 helpers ----------
__device__ __forceinline__ u16 f2bf(float f) {
  union { float f; u32 u; } x; x.f = f;
  u32 r = x.u + 0x7fffu + ((x.u >> 16) & 1u);   // round-to-nearest-even
  return (u16)(r >> 16);
}

// two 16B chunks -> one v16bf fragment (global or LDS pointers)
union FragU { uint4 q[2]; v16bf h; };
__device__ __forceinline__ v16bf ldfrag(const u16* p0, const u16* p1) {
  FragU f;
  f.q[0] = *(const uint4*)p0;
  f.q[1] = *(const uint4*)p1;
  return f.h;
}

#define WMMA_BF16(acc, a, b)                                                   \
  acc = __builtin_amdgcn_wmma_f32_16x16x32_bf16(false, a, false, b, (short)0,  \
                                                acc, false, false)
#define V8F_ZERO v8f{0.f,0.f,0.f,0.f,0.f,0.f,0.f,0.f}

// ---------- Tensor Data Mover: 2-D tile (2-byte elems) global -> LDS ----------
__device__ __forceinline__ void tdm_load_2d(const void* gaddr, u32 lds_off,
                                            u32 dim0, u32 dim1, u32 stride0,
                                            u32 tile0, u32 tile1) {
  u64 ga = (u64)(size_t)gaddr;
  tdm_vu4 g0;
  g0[0] = 1u;                                            // count=1 (user D#)
  g0[1] = lds_off;                                       // lds_addr
  g0[2] = (u32)ga;                                       // global_addr[31:0]
  g0[3] = (u32)((ga >> 32) & 0x01ffffffu) | (2u << 30);  // addr[56:32] | type=2
  tdm_vi8 g1;
  g1[0] = (int)(1u << 16);                               // data_size = 2 bytes
  g1[1] = (int)((dim0 & 0xffffu) << 16);                 // tensor_dim0 lo16
  g1[2] = (int)((dim0 >> 16) | ((dim1 & 0xffffu) << 16));// dim0 hi / dim1 lo
  g1[3] = (int)((dim1 >> 16) | (tile0 << 16));           // dim1 hi / tile_dim0
  g1[4] = (int)(tile1 & 0xffffu);                        // tile_dim1 (tile2=0)
  g1[5] = (int)stride0;                                  // dim0_stride[31:0]
  g1[6] = 0;
  g1[7] = 0;
  tdm_vi4 gz = {0, 0, 0, 0};
#if defined(__clang_major__) && (__clang_major__ >= 23)
  tdm_vi8 gz8 = {0, 0, 0, 0, 0, 0, 0, 0};
  __builtin_amdgcn_tensor_load_to_lds(g0, g1, gz, gz, gz8, 0);
#else
  __builtin_amdgcn_tensor_load_to_lds(g0, g1, gz, gz, 0);
#endif
}

// =====================================================================
// conv1 (+pool+relu): f32 [B,16,3750] -> bf16 TOKEN-MAJOR [B,1874,256]
// =====================================================================
__global__ void k_conv1(const float* __restrict__ x, const float* __restrict__ w,
                        const float* __restrict__ bias, u16* __restrict__ out) {
  int idx = blockIdx.x * blockDim.x + threadIdx.x;
  if (idx >= Bn * H * L1) return;
  int jp = idx % L1; int t = idx / L1; int o = t % H; int b = t / H;
  const float* xb = x + (size_t)b * CIN * L0;
  const float* wo = w + o * CIN * 2;
  float m = -3.0e38f;
  #pragma unroll
  for (int s = 0; s < 2; ++s) {
    int l = 2 * jp + s;
    float acc = 0.f;
    #pragma unroll
    for (int c = 0; c < CIN; ++c) {
      const float* xc = xb + c * L0 + l;
      acc += xc[0] * wo[c * 2 + 0] + xc[1] * wo[c * 2 + 1];
    }
    m = fmaxf(m, acc);
  }
  out[((size_t)b * L1 + jp) * H + o] = f2bf(fmaxf(m + bias[o], 0.f));
}

// ---------- f32 [K,N] -> bf16 transposed [N,K] ----------
__global__ void k_castT(const float* __restrict__ w, u16* __restrict__ wt,
                        int K, int N) {
  int i = blockIdx.x * blockDim.x + threadIdx.x;
  if (i >= K * N) return;
  int k = i % K, n = i / K;
  wt[i] = f2bf(w[(size_t)k * N + n]);
}

// ---------- conv weight repack: [O,C,2] f32 -> [O, t*C+c] bf16 ----------
__global__ void k_repack_convw(const float* __restrict__ w, u16* __restrict__ wt,
                               int C) {
  int i = blockIdx.x * blockDim.x + threadIdx.x;
  if (i >= H * C * 2) return;
  int c = i % C; int t2 = i / C; int t = t2 % 2; int o = t2 / 2;
  wt[(size_t)o * 2 * C + t * C + c] = f2bf(w[((size_t)o * C + c) * 2 + t]);
}

// =====================================================================
// conv2/conv3: implicit-im2col WMMA GEMM with k = t*C + c ordering.
// actT bf16 token-major [B,Lin,C]; wt bf16 [O=256, 2C].
// Fused bias+maxpool2+relu; token-major bf16 out (+ optional f32).
// A-fragment = two contiguous b128 loads (chunks never cross the t boundary).
// =====================================================================
__global__ void __launch_bounds__(32)
k_conv_gemm(const u16* __restrict__ actT, const u16* __restrict__ wt,
            const float* __restrict__ bias, u16* __restrict__ outb,
            float* __restrict__ outf, int C, int Lin) {
  int lane = threadIdx.x & 31, lo = lane & 15, hi = lane >> 4;
  int mt = blockIdx.x, nt = blockIdx.y, b = blockIdx.z;
  int Lconv = Lin - 1, Lp = Lconv / 2, K = 2 * C;
  int row = mt * 16 + lo; if (row > Lconv - 1) row = Lconv - 1;
  const u16* abase = actT + (size_t)b * Lin * C;
  v8f acc[4];
  #pragma unroll
  for (int s = 0; s < 4; ++s) acc[s] = V8F_ZERO;
  for (int k0 = 0; k0 < K; k0 += 32) {
    int kc0 = k0 + hi * 8;            // chunk starts, multiples of 8
    int kc1 = k0 + 16 + hi * 8;
    int t0 = (kc0 >= C) ? 1 : 0, c0 = kc0 - t0 * C;
    int t1 = (kc1 >= C) ? 1 : 0, c1 = kc1 - t1 * C;
    FragU fa;
    fa.q[0] = *(const uint4*)(abase + (size_t)(row + t0) * C + c0);
    fa.q[1] = *(const uint4*)(abase + (size_t)(row + t1) * C + c1);
    v16bf af = fa.h;
    #pragma unroll
    for (int s = 0; s < 4; ++s) {
      int col = nt * 64 + s * 16 + lo;
      const u16* bp = wt + (size_t)col * K + k0 + hi * 8;
      v16bf bf = ldfrag(bp, bp + 16);
      WMMA_BF16(acc[s], af, bf);
    }
  }
  #pragma unroll
  for (int s = 0; s < 4; ++s) {
    int col = nt * 64 + s * 16 + lo;
    float bcol = bias[col];
    #pragma unroll
    for (int j = 0; j < 4; ++j) {      // maxpool pairs live within one lane
      float pm = fmaxf(acc[s][2 * j], acc[s][2 * j + 1]);
      int jp = mt * 8 + hi * 4 + j;
      float v = fmaxf(pm + bcol, 0.f);
      if (jp < Lp) {
        size_t di = ((size_t)b * Lp + jp) * H + col;
        outb[di] = f2bf(v);
        if (outf) outf[di] = v;
      }
    }
  }
}

// =====================================================================
// Generic GEMM: C[M,N] = A[M,K]bf16 * WT[N,K]bf16 + bias, 32x64 tile/wave.
// Weight panel staged into LDS by the Tensor Data Mover; 8 WMMA / k-step.
// =====================================================================
__global__ void __launch_bounds__(32)
k_gemm(const u16* __restrict__ A, const u16* __restrict__ WT,
       const float* __restrict__ bias, float* __restrict__ Cf,
       u16* __restrict__ Cb, int M, int N, int K, int relu) {
  __shared__ __align__(16) u16 Bs[64 * 256];      // 32 KB max (K<=256)
  int lane = threadIdx.x & 31, lo = lane & 15, hi = lane >> 4;
  int mt = blockIdx.x, nt = blockIdx.y;
  int row0 = mt * 32 + lo;      if (row0 >= M) row0 = M - 1;
  int row1 = mt * 32 + 16 + lo; if (row1 >= M) row1 = M - 1;
  int colbase = nt * 64;
  tdm_load_2d(WT + (size_t)colbase * K, (u32)(size_t)(void*)Bs,
              (u32)K, (u32)N, (u32)K, (u32)K, 64u);
  const u16* Ar0 = A + (size_t)row0 * K;
  const u16* Ar1 = A + (size_t)row1 * K;
  v8f acc[2][4];
  #pragma unroll
  for (int mi = 0; mi < 2; ++mi)
    #pragma unroll
    for (int s = 0; s < 4; ++s) acc[mi][s] = V8F_ZERO;
  __builtin_amdgcn_s_wait_tensorcnt(0);
  for (int k0 = 0; k0 < K; k0 += 32) {
    if (k0 + 32 < K) {
      __builtin_prefetch(Ar0 + k0 + 32, 0, 1);
      __builtin_prefetch(Ar1 + k0 + 32, 0, 1);
    }
    v16bf af0 = ldfrag(Ar0 + k0 + hi * 8, Ar0 + k0 + 16 + hi * 8);
    v16bf af1 = ldfrag(Ar1 + k0 + hi * 8, Ar1 + k0 + 16 + hi * 8);
    #pragma unroll
    for (int s = 0; s < 4; ++s) {
      const u16* bp = Bs + (size_t)(s * 16 + lo) * K + k0 + hi * 8;
      v16bf bf = ldfrag(bp, bp + 16);
      WMMA_BF16(acc[0][s], af0, bf);
      WMMA_BF16(acc[1][s], af1, bf);
    }
  }
  #pragma unroll
  for (int mi = 0; mi < 2; ++mi) {
    #pragma unroll
    for (int s = 0; s < 4; ++s) {
      int col = colbase + s * 16 + lo;
      float bcol = bias[col];
      #pragma unroll
      for (int r = 0; r < 8; ++r) {
        int m = mt * 32 + mi * 16 + r + hi * 8;
        if (m < M) {
          float v = acc[mi][s][r] + bcol;
          if (relu) v = fmaxf(v, 0.f);
          size_t di = (size_t)m * N + col;
          if (Cf) Cf[di] = v;
          if (Cb) Cb[di] = f2bf(v);
        }
      }
    }
  }
}

// =====================================================================
// scores[h,q,k] = Q.K^T/sqrt(32) — 32(q) x 64(k) per wave, K=32
// =====================================================================
__global__ void __launch_bounds__(32)
k_scores(const u16* __restrict__ q, const u16* __restrict__ kmat,
         float* __restrict__ sc, int b) {
  int lane = threadIdx.x & 31, lo = lane & 15, hi = lane >> 4;
  int qt = blockIdx.x, ktg = blockIdx.y, h = blockIdx.z;
  int qp0 = qt * 32 + lo;      if (qp0 >= LSEQ) qp0 = LSEQ - 1;
  int qp1 = qt * 32 + 16 + lo; if (qp1 >= LSEQ) qp1 = LSEQ - 1;
  const u16* qr0 = q + (((size_t)b * LSEQ + qp0) * NH + h) * DK;
  const u16* qr1 = q + (((size_t)b * LSEQ + qp1) * NH + h) * DK;
  v16bf af0 = ldfrag(qr0 + hi * 8, qr0 + 16 + hi * 8);
  v16bf af1 = ldfrag(qr1 + hi * 8, qr1 + 16 + hi * 8);
  const float scale = 0.17677669529663687f;   // 1/sqrt(32)
  #pragma unroll
  for (int s = 0; s < 4; ++s) {
    int kt = ktg * 4 + s;
    int kp = kt * 16 + lo; if (kp >= LSEQ) kp = LSEQ - 1;
    const u16* krow = kmat + (((size_t)b * LSEQ + kp) * NH + h) * DK + hi * 8;
    v16bf bf = ldfrag(krow, krow + 16);
    v8f acc0 = V8F_ZERO, acc1 = V8F_ZERO;
    WMMA_BF16(acc0, af0, bf);
    WMMA_BF16(acc1, af1, bf);
    #pragma unroll
    for (int r = 0; r < 8; ++r) {
      int n = kt * 16 + lo;
      int m0 = qt * 32 + r + hi * 8;
      int m1 = m0 + 16;
      if (m0 < LSEQ && n < LSEQ)
        sc[((size_t)h * LSEQ + m0) * SPAD + n] = acc0[r] * scale;
      if (m1 < LSEQ && n < LSEQ)
        sc[((size_t)h * LSEQ + m1) * SPAD + n] = acc1[r] * scale;
    }
  }
}

// ---------- softmax: one wave per (h,q) row; zero-fills key padding ----------
__global__ void k_softmax(const float* __restrict__ sc, u16* __restrict__ attn) {
  int row = blockIdx.x;                 // 0 .. NH*LSEQ-1
  int lane = threadIdx.x & 31;
  const float* srow = sc + (size_t)row * SPAD;
  u16* arow = attn + (size_t)row * SPAD;
  float mx = -3.0e38f;
  for (int i = lane; i < LSEQ; i += 32) mx = fmaxf(mx, srow[i]);
  for (int off = 16; off > 0; off >>= 1) mx = fmaxf(mx, __shfl_xor(mx, off, 32));
  float sum = 0.f;
  for (int i = lane; i < LSEQ; i += 32) sum += __expf(srow[i] - mx);
  for (int off = 16; off > 0; off >>= 1) sum += __shfl_xor(sum, off, 32);
  float inv = 1.f / sum;
  for (int i = lane; i < SPAD; i += 32)
    arow[i] = f2bf((i < LSEQ) ? __expf(srow[i] - mx) * inv : 0.f);
}

// ---------- transposed, key-padded V: vt[b,h,d,kp] (zeros in padding) ----------
__global__ void k_vtrans(const u16* __restrict__ vb, u16* __restrict__ vt) {
  int i = blockIdx.x * blockDim.x + threadIdx.x;
  if (i >= Bn * NH * DV * SPAD) return;
  int kp = i % SPAD; int t = i / SPAD;
  int d = t % DV; t /= DV;
  int h = t % NH; int b = t / NH;
  vt[i] = (kp < LSEQ) ? vb[(((size_t)b * LSEQ + kp) * NH + h) * DV + d] : (u16)0;
}

// ---------- O = attn @ V : K=480 padded, 32(q) x 32(d) per wave ----------
__global__ void __launch_bounds__(32)
k_attnv(const u16* __restrict__ attn, const u16* __restrict__ vt,
        u16* __restrict__ outb, int b) {
  int lane = threadIdx.x & 31, lo = lane & 15, hi = lane >> 4;
  int qt = blockIdx.x, h = blockIdx.y;
  int qp0 = qt * 32 + lo;      if (qp0 >= LSEQ) qp0 = LSEQ - 1;
  int qp1 = qt * 32 + 16 + lo; if (qp1 >= LSEQ) qp1 = LSEQ - 1;
  const u16* ar0 = attn + ((size_t)h * LSEQ + qp0) * SPAD;
  const u16* ar1 = attn + ((size_t)h * LSEQ + qp1) * SPAD;
  v8f acc[2][2];
  #pragma unroll
  for (int mi = 0; mi < 2; ++mi)
    #pragma unroll
    for (int s = 0; s < 2; ++s) acc[mi][s] = V8F_ZERO;
  for (int k0 = 0; k0 < SPAD; k0 += 32) {
    v16bf af0 = ldfrag(ar0 + k0 + hi * 8, ar0 + k0 + 16 + hi * 8);
    v16bf af1 = ldfrag(ar1 + k0 + hi * 8, ar1 + k0 + 16 + hi * 8);
    #pragma unroll
    for (int s = 0; s < 2; ++s) {
      int d = s * 16 + lo;
      const u16* bp = vt + (((size_t)b * NH + h) * DV + d) * SPAD + k0 + hi * 8;
      v16bf bf = ldfrag(bp, bp + 16);
      WMMA_BF16(acc[0][s], af0, bf);
      WMMA_BF16(acc[1][s], af1, bf);
    }
  }
  #pragma unroll
  for (int mi = 0; mi < 2; ++mi) {
    #pragma unroll
    for (int s = 0; s < 2; ++s) {
      #pragma unroll
      for (int r = 0; r < 8; ++r) {
        int m = qt * 32 + mi * 16 + r + hi * 8;
        if (m < LSEQ)
          outb[((size_t)b * LSEQ + m) * (NH * DV) + h * DV + s * 16 + lo] =
              f2bf(acc[mi][s][r]);
      }
    }
  }
}

// ---------- residual + LayerNorm (row = token, wave32, 8 elems/lane) ----------
__global__ void k_resid_ln(const float* __restrict__ y, float* __restrict__ data,
                           u16* __restrict__ datab, const float* __restrict__ g,
                           const float* __restrict__ bta) {
  int row = blockIdx.x;
  int lane = threadIdx.x & 31;
  const float* yr = y + (size_t)row * H;
  float* dr = data + (size_t)row * H;
  u16* db = datab + (size_t)row * H;
  float vals[8], s = 0.f, s2 = 0.f;
  #pragma unroll
  for (int i = 0; i < 8; ++i) {
    int c = lane + 32 * i;
    float t = yr[c] + dr[c];
    vals[i] = t; s += t; s2 += t * t;
  }
  for (int off = 16; off > 0; off >>= 1) {
    s += __shfl_xor(s, off, 32); s2 += __shfl_xor(s2, off, 32);
  }
  float mu = s * (1.f / H);
  float var = s2 * (1.f / H) - mu * mu;
  float inv = rsqrtf(var + 1e-5f);
  #pragma unroll
  for (int i = 0; i < 8; ++i) {
    int c = lane + 32 * i;
    float o = (vals[i] - mu) * inv * g[c] + bta[c];
    dr[c] = o; db[c] = f2bf(o);
  }
}

// ---------- sum over sequence ----------
__global__ void k_seqsum(const float* __restrict__ data, float* __restrict__ pooled) {
  int i = blockIdx.x * blockDim.x + threadIdx.x;
  if (i >= Bn * H) return;
  int c = i % H, b = i / H;
  float s = 0.f;
  for (int l = 0; l < LSEQ; ++l) s += data[((size_t)b * LSEQ + l) * H + c];
  pooled[i] = s;
}

// ---------- tiny dense head (f32) ----------
__global__ void k_dense(const float* __restrict__ X, const float* __restrict__ W,
                        const float* __restrict__ bias, float* __restrict__ Y,
                        int M, int N, int K, int relu) {
  int i = blockIdx.x * blockDim.x + threadIdx.x;
  if (i >= M * N) return;
  int n = i % N, m = i / N;
  float s = bias[n];
  for (int k = 0; k < K; ++k) s += X[m * K + k] * W[k * N + n];
  Y[i] = relu ? fmaxf(s, 0.f) : s;
}

// =====================================================================
// host orchestration
// =====================================================================
static char* ws_take(char*& p, size_t bytes) {
  char* r = p;
  p += (bytes + 255) & ~(size_t)255;
  return r;
}

extern "C" void kernel_launch(void* const* d_in, const int* in_sizes, int n_in,
                              void* d_out, int out_size, void* d_ws, size_t ws_size,
                              hipStream_t stream) {
  (void)in_sizes; (void)n_in; (void)out_size; (void)ws_size;
  const float* x       = (const float*)d_in[0];
  const float* conv1_w = (const float*)d_in[1];
  const float* conv1_b = (const float*)d_in[2];
  const float* conv2_w = (const float*)d_in[3];
  const float* conv2_b = (const float*)d_in[4];
  const float* conv3_w = (const float*)d_in[5];
  const float* conv3_b = (const float*)d_in[6];
  const float* h1_w = (const float*)d_in[55];
  const float* h1_b = (const float*)d_in[56];
  const float* h2_w = (const float*)d_in[57];
  const float* h2_b = (const float*)d_in[58];
  const float* h3_w = (const float*)d_in[59];
  const float* h3_b = (const float*)d_in[60];

  // ---- workspace carve ----
  char* p = (char*)d_ws;
  u16*   a1t    = (u16*)  ws_take(p, (size_t)Bn * L1 * H * 2);   // token-major
  u16*   a2t    = (u16*)  ws_take(p, (size_t)Bn * L2 * H * 2);   // token-major
  float* dataf  = (float*)ws_take(p, (size_t)BL * H * 4);
  u16*   datab  = (u16*)  ws_take(p, (size_t)BL * H * 2);
  u16*   qb     = (u16*)  ws_take(p, (size_t)BL * H * 2);
  u16*   kb     = (u16*)  ws_take(p, (size_t)BL * H * 2);
  u16*   vb     = (u16*)  ws_take(p, (size_t)BL * H * 2);
  u16*   vt     = (u16*)  ws_take(p, (size_t)Bn * NH * DV * SPAD * 2);
  float* scores = (float*)ws_take(p, (size_t)NH * LSEQ * SPAD * 4);
  u16*   attn   = (u16*)  ws_take(p, (size_t)NH * LSEQ * SPAD * 2);
  u16*   aout   = (u16*)  ws_take(p, (size_t)BL * H * 2);
  float* tmpf   = (float*)ws_take(p, (size_t)BL * H * 4);
  u16*   hid    = (u16*)  ws_take(p, (size_t)BL * INNER * 2);
  u16*   cw2    = (u16*)  ws_take(p, (size_t)H * 2 * H * 2);
  u16*   cw3    = (u16*)  ws_take(p, (size_t)H * 2 * H * 2);
  u16*   wqb    = (u16*)  ws_take(p, (size_t)H * H * 2);
  u16*   wkb    = (u16*)  ws_take(p, (size_t)H * H * 2);
  u16*   wvb    = (u16*)  ws_take(p, (size_t)H * H * 2);
  u16*   fcb    = (u16*)  ws_take(p, (size_t)H * H * 2);
  u16*   w1b    = (u16*)  ws_take(p, (size_t)H * INNER * 2);
  u16*   w2b    = (u16*)  ws_take(p, (size_t)INNER * H * 2);
  float* pooled = (float*)ws_take(p, (size_t)Bn * H * 4);
  float* t1     = (float*)ws_take(p, (size_t)Bn * 200 * 4);
  float* t2     = (float*)ws_take(p, (size_t)Bn * 200 * 4);

  auto castT = [&](const float* s, u16* d, int K, int N) {  // [K,N] -> [N,K] bf16
    k_castT<<<(K * N + 255) / 256, 256, 0, stream>>>(s, d, K, N);
  };

  // ---- conv front-end (token-major activations, k = t*C + c im2col) ----
  {
    int tot = Bn * H * L1;
    k_conv1<<<(tot + 255) / 256, 256, 0, stream>>>(x, conv1_w, conv1_b, a1t);
  }
  k_repack_convw<<<(H * H * 2 + 255) / 256, 256, 0, stream>>>(conv2_w, cw2, H);
  k_repack_convw<<<(H * H * 2 + 255) / 256, 256, 0, stream>>>(conv3_w, cw3, H);
  k_conv_gemm<<<dim3(117, 4, Bn), 32, 0, stream>>>(a1t, cw2, conv2_b,
                                                   a2t, nullptr, H, L1);
  k_conv_gemm<<<dim3(59, 4, Bn), 32, 0, stream>>>(a2t, cw3, conv3_b,
                                                  datab, dataf, H, L2);

  const int MT32 = (BL + 31) / 32;    // 467 (exact)
  const int QT32 = (LSEQ + 31) / 32;  // 15

  // ---- 3 encoder blocks (layer0, layer1, task) ----
  for (int li = 0; li < 3; ++li) {
    int pb = 7 + 16 * li;
    const float* wq_w = (const float*)d_in[pb + 0];
    const float* wq_b = (const float*)d_in[pb + 1];
    const float* wk_w = (const float*)d_in[pb + 2];
    const float* wk_b = (const float*)d_in[pb + 3];
    const float* wv_w = (const float*)d_in[pb + 4];
    const float* wv_b = (const float*)d_in[pb + 5];
    const float* fc_w = (const float*)d_in[pb + 6];
    const float* fc_b = (const float*)d_in[pb + 7];
    const float* ln1g = (const float*)d_in[pb + 8];
    const float* ln1b = (const float*)d_in[pb + 9];
    const float* w1   = (const float*)d_in[pb + 10];
    const float* b1   = (const float*)d_in[pb + 11];
    const float* w2   = (const float*)d_in[pb + 12];
    const float* b2   = (const float*)d_in[pb + 13];
    const float* ln2g = (const float*)d_in[pb + 14];
    const float* ln2b = (const float*)d_in[pb + 15];

    castT(wq_w, wqb, H, H); castT(wk_w, wkb, H, H); castT(wv_w, wvb, H, H);
    castT(fc_w, fcb, H, H); castT(w1, w1b, H, INNER); castT(w2, w2b, INNER, H);

    // QKV projections (32x64 tiles, TDM-staged weight panels)
    k_gemm<<<dim3(MT32, 4), 32, 0, stream>>>(datab, wqb, wq_b, nullptr, qb, BL, H, H, 0);
    k_gemm<<<dim3(MT32, 4), 32, 0, stream>>>(datab, wkb, wk_b, nullptr, kb, BL, H, H, 0);
    k_gemm<<<dim3(MT32, 4), 32, 0, stream>>>(datab, wvb, wv_b, nullptr, vb, BL, H, H, 0);
    k_vtrans<<<(Bn * NH * DV * SPAD + 255) / 256, 256, 0, stream>>>(vb, vt);

    // attention, one batch at a time (bounds score scratch)
    for (int b = 0; b < Bn; ++b) {
      k_scores<<<dim3(QT32, 8, NH), 32, 0, stream>>>(qb, kb, scores, b);
      k_softmax<<<dim3(NH * LSEQ), 32, 0, stream>>>(scores, attn);
      k_attnv<<<dim3(QT32, NH), 32, 0, stream>>>(attn, vt, aout, b);
    }

    // output projection + residual + LN1
    k_gemm<<<dim3(MT32, 4), 32, 0, stream>>>(aout, fcb, fc_b, tmpf, nullptr, BL, H, H, 0);
    k_resid_ln<<<dim3(BL), 32, 0, stream>>>(tmpf, dataf, datab, ln1g, ln1b);

    // FFN + residual + LN2
    k_gemm<<<dim3(MT32, 2), 32, 0, stream>>>(datab, w1b, b1, nullptr, hid, BL, INNER, H, 1);
    k_gemm<<<dim3(MT32, 4), 32, 0, stream>>>(hid, w2b, b2, tmpf, nullptr, BL, H, INNER, 0);
    k_resid_ln<<<dim3(BL), 32, 0, stream>>>(tmpf, dataf, datab, ln2g, ln2b);
  }

  // ---- head ----
  k_seqsum<<<(Bn * H + 255) / 256, 256, 0, stream>>>(dataf, pooled);
  k_dense<<<(Bn * 200 + 255) / 256, 256, 0, stream>>>(pooled, h1_w, h1_b, t1, Bn, 200, H, 1);
  k_dense<<<(Bn * 200 + 255) / 256, 256, 0, stream>>>(t1, h2_w, h2_b, t2, Bn, 200, 200, 1);
  k_dense<<<(Bn * OUTD + 255) / 256, 256, 0, stream>>>(t2, h3_w, h3_b, (float*)d_out,
                                                       Bn, OUTD, 200, 0);
}
